// PAM_79456894976424
// MI455X (gfx1250) — matrix operand order repaired and checked by reading, not compiled, not run on hardware
//
#include <hip/hip_runtime.h>
#include <hip/hip_bf16.h>

// PAM (position attention) for x[1,84,112,512]:
//   q=xWb k=xWc v=xWd ; S=q k^T ; A=softmax(S) ; out = gamma*(A v) + x
// hw = 9408 = 588*16, C = 512, d = 64.
// Phase 2 = flash attention, key-parallel online softmax, TWO key tiles per
// iteration so P@V WMMAs run with a full K=32 (no zero padding).
// Q tile staged into LDS once per block via global_load_async_to_lds_b64.

#define HW      9408
#define CCH     512
#define NT_J    588            // 9408/16 key tiles
#define NT_PROJ 40             // 4 (Wb) + 4 (Wc) + 32 (Wd) column tiles
#define NPAIR   74             // 147 tiles/group = 73 pairs + 1 singleton

typedef __attribute__((ext_vector_type(16))) __bf16 v16bf;
typedef __attribute__((ext_vector_type(8)))  float  v8f;

union BfVec { v16bf v; __bf16 e[16]; unsigned short s[16]; };
union F8    { v8f  v; float f[8]; };

__device__ inline unsigned short bf_bits(float f) {
  union { __bf16 b; unsigned short s; } u; u.b = (__bf16)f; return u.s;
}

// ---------------------------------------------------------------------------
// Phase 1: projections.  One wave per 16x16 output tile of [9408 x 640]
// (Q cols 0..63 | K cols 0..63 | V cols 0..511).  fp32 -> bf16 with packed
// hardware converts, accumulate fp32 via v_wmma_f32_16x16x32_bf16.
// V is stored TRANSPOSED ([512 x 9408]) so phase-2 B loads are contiguous.
// ---------------------------------------------------------------------------
__global__ __launch_bounds__(256)
void pam_proj_kernel(const float* __restrict__ x,
                     const float* __restrict__ Wb,
                     const float* __restrict__ Wc,
                     const float* __restrict__ Wd,
                     unsigned short* __restrict__ qb,
                     unsigned short* __restrict__ kbuf,
                     unsigned short* __restrict__ vbT)
{
  const int lane = threadIdx.x & 31;
  const int wv   = threadIdx.x >> 5;
  const int gw   = blockIdx.x * 8 + wv;     // 2940*8 == 588*40 exactly
  const int mt   = gw / NT_PROJ;
  const int nt   = gw % NT_PROJ;
  const int q0   = mt * 16;
  const int lh   = lane >> 4;
  const int nl   = lane & 15;

  const float* W; int ld, ncol;
  if (nt < 4)      { W = Wb; ld = 64;  ncol = nt * 16;       }
  else if (nt < 8) { W = Wc; ld = 64;  ncol = (nt - 4) * 16; }
  else             { W = Wd; ld = 512; ncol = (nt - 8) * 16; }

  const float* xrow = x + (size_t)(q0 + nl) * CCH;   // A row for this lane

  F8 accu;
#pragma unroll
  for (int i = 0; i < 8; i++) accu.f[i] = 0.f;
  v8f acc = accu.v;

  for (int kb = 0; kb < CCH; kb += 32) {
    // A operand (16x32 bf16): lane half selects contiguous K (k = 16*lh + s).
    BfVec a;
    const float* p0 = xrow + kb + 16 * lh;
    const float* p1 = xrow + kb + 16 * lh + 8;
#pragma unroll
    for (int i = 0; i < 8; i++) { a.e[i] = (__bf16)p0[i]; a.e[8 + i] = (__bf16)p1[i]; }

    // B operand (32x16 bf16): lane nl = output column, halves split K.
    BfVec b;
    const int k0 = kb + (lh << 4);
#pragma unroll
    for (int e = 0; e < 16; e++) b.e[e] = (__bf16)W[(size_t)(k0 + e) * ld + ncol + nl];

    acc = __builtin_amdgcn_wmma_f32_16x16x32_bf16(false, a.v, false, b.v,
                                                  (short)0, acc, false, false);
  }

  F8 res; res.v = acc;
  if (nt < 8) {                         // Q / K: row-major [9408 x 64]
    unsigned short* dst = (nt < 4) ? qb : kbuf;
    const int c0 = (nt & 3) * 16;
#pragma unroll
    for (int r = 0; r < 8; r++)
      dst[(size_t)(q0 + r + 8 * lh) * 64 + c0 + nl] = bf_bits(res.f[r]);
  } else {                              // V transposed: [512 x 9408]
    const int vc = (nt - 8) * 16 + nl;
#pragma unroll
    for (int r = 0; r < 8; r++)
      vbT[(size_t)vc * HW + q0 + r + 8 * lh] = bf_bits(res.f[r]);
  }
}

// ---------------------------------------------------------------------------
// Phase 2: key-parallel flash attention.  1 block = 16 query rows, 8 waves.
// Wave (g,h): g = key group (tiles j ≡ g mod 4), h = column half (256 cols).
// Each wave consumes key tiles in PAIRS (j0 = g+8u, j1 = j0+4):
//   A = [P_j0 | P_j1] (16x32), B = [V_j0 ; V_j1] (32x16)  -> full-K WMMAs.
// No barriers in the main loop; partials merged once at the end (ds_add_f32).
// ---------------------------------------------------------------------------
__global__ __launch_bounds__(256)
void pam_attn_kernel(const float* __restrict__ x,
                     const unsigned short* __restrict__ qb,
                     const unsigned short* __restrict__ kbuf,
                     const unsigned short* __restrict__ vbT,
                     const float* __restrict__ gamma,
                     float* __restrict__ out)
{
  __shared__ float ctx_s[16 * CCH];          // merged softmax-weighted context
  __shared__ float m_s[4][16];               // per-group row maxima
  __shared__ float l_s[4][16];               // per-group row sums
  __shared__ unsigned short P_s[8][16 * 32]; // per-wave [P_j0 | P_j1] scratch
  __shared__ __align__(16) unsigned short q_s[16 * 64];  // Q tile (2 KB)

  const int lane = threadIdx.x & 31;
  const int wv   = threadIdx.x >> 5;
  const int g    = wv & 3;                   // key group
  const int h    = wv >> 2;                  // column half
  const int q0   = blockIdx.x * 16;
  const int lh   = lane >> 4;
  const int nl   = lane & 15;
  const int cb   = h * 256;

  // Stage the Q tile into LDS once for the whole block: one async 8-byte
  // chunk per thread (ASYNCcnt-tracked, §15.18 GLOBAL_LOAD_ASYNC_TO_LDS_B64).
  {
    const unsigned short* gsrc = qb + (size_t)q0 * 64 + threadIdx.x * 4;
    unsigned lds_dst = (unsigned)(uintptr_t)(&q_s[threadIdx.x * 4]);
    asm volatile("global_load_async_to_lds_b64 %0, %1, off"
                 :: "v"(lds_dst), "v"((unsigned long long)(uintptr_t)gsrc)
                 : "memory");
    asm volatile("s_wait_asynccnt 0" ::: "memory");
  }

  for (int i = threadIdx.x; i < 16 * CCH; i += 256) ctx_s[i] = 0.f;
  __syncthreads();                           // q_s visible to all waves

  F8 z;
#pragma unroll
  for (int i = 0; i < 8; i++) z.f[i] = 0.f;
  v8f acc[16];
#pragma unroll
  for (int t = 0; t < 16; t++) acc[t] = z.v;

  // Q tile as A operand, K-dim 0..63 (built from LDS)
  v16bf qa0, qa1;
  {
    const unsigned short* qrow = q_s + nl * 64;
    BfVec a;
#pragma unroll
    for (int i = 0; i < 8; i++) { a.s[i] = qrow[16 * lh + i]; a.s[8 + i] = qrow[16 * lh + 8 + i]; }
    qa0 = a.v;
#pragma unroll
    for (int i = 0; i < 8; i++) { a.s[i] = qrow[32 + 16 * lh + i]; a.s[8 + i] = qrow[40 + 16 * lh + i]; }
    qa1 = a.v;
  }

  float mst[8], lst[8];
#pragma unroll
  for (int r = 0; r < 8; r++) { mst[r] = -3.402823466e38f; lst[r] = 0.f; }

  for (int u = 0; u < NPAIR; u++) {
    const int  j0    = g + 8 * u;
    const int  j1    = j0 + 4;
    const bool have2 = (u < NPAIR - 1);      // last iteration: phantom 2nd tile

    // ---- S = Q K^T for both tiles (B = K^T, contiguous row loads) ----------
    const unsigned short* kr0 = kbuf + (size_t)(j0 * 16 + nl) * 64 + (lh << 4);
    BfVec b0, b1, b2, b3;
#pragma unroll
    for (int i = 0; i < 16; i++) { b0.s[i] = kr0[i]; b1.s[i] = kr0[32 + i]; }
    if (have2) {
      const unsigned short* kr1 = kbuf + (size_t)(j1 * 16 + nl) * 64 + (lh << 4);
#pragma unroll
      for (int i = 0; i < 16; i++) { b2.s[i] = kr1[i]; b3.s[i] = kr1[32 + i]; }
    } else {
#pragma unroll
      for (int i = 0; i < 16; i++) { b2.s[i] = 0; b3.s[i] = 0; }
    }

    v8f s1 = z.v, s2 = z.v;
    s1 = __builtin_amdgcn_wmma_f32_16x16x32_bf16(false, qa0, false, b0.v, (short)0, s1, false, false);
    s1 = __builtin_amdgcn_wmma_f32_16x16x32_bf16(false, qa1, false, b1.v, (short)0, s1, false, false);
    s2 = __builtin_amdgcn_wmma_f32_16x16x32_bf16(false, qa0, false, b2.v, (short)0, s2, false, false);
    s2 = __builtin_amdgcn_wmma_f32_16x16x32_bf16(false, qa1, false, b3.v, (short)0, s2, false, false);

    // ---- joint online softmax over the 16x32 score strip -------------------
    F8 sv1, sv2; sv1.v = s1; sv2.v = s2;
    float scr[8];
#pragma unroll
    for (int r = 0; r < 8; r++) {
      float v1 = sv1.f[r];
      float v2 = have2 ? sv2.f[r] : -3.402823466e38f;   // mask phantom tile
      float rm = fmaxf(v1, v2);
      rm = fmaxf(rm, __shfl_xor(rm, 1, 32));
      rm = fmaxf(rm, __shfl_xor(rm, 2, 32));
      rm = fmaxf(rm, __shfl_xor(rm, 4, 32));
      rm = fmaxf(rm, __shfl_xor(rm, 8, 32));
      float mnew = fmaxf(mst[r], rm);
      float sc   = __expf(mst[r] - mnew);
      float p1   = __expf(v1 - mnew);
      float p2   = __expf(v2 - mnew);
      float rs   = p1 + p2;
      rs += __shfl_xor(rs, 1, 32);
      rs += __shfl_xor(rs, 2, 32);
      rs += __shfl_xor(rs, 4, 32);
      rs += __shfl_xor(rs, 8, 32);
      lst[r] = lst[r] * sc + rs;
      mst[r] = mnew;
      scr[r] = sc;
      const int row = r + 8 * lh;
      P_s[wv][row * 32 + nl]      = bf_bits(p1);   // same-wave LDS: in-order
      P_s[wv][row * 32 + 16 + nl] = bf_bits(p2);
    }

    // A = [P_j0 | P_j1] (16x32) re-laid-out to A-operand layout via LDS.
    BfVec ap;
    {
      const unsigned short* pr = &P_s[wv][nl * 32 + 16 * lh];
#pragma unroll
      for (int i = 0; i < 8; i++) { ap.s[i] = pr[i]; ap.s[8 + i] = pr[8 + i]; }
    }

    // Rescale accumulators only when the running max actually moved.
    bool ns = false;
#pragma unroll
    for (int r = 0; r < 8; r++) ns |= (scr[r] != 1.0f);
    if (__any((int)ns)) {
#pragma unroll
      for (int nt = 0; nt < 16; nt++) {
        F8 ta; ta.v = acc[nt];
#pragma unroll
        for (int r = 0; r < 8; r++) ta.f[r] *= scr[r];
        acc[nt] = ta.v;
      }
    }

    // ---- [P_j0|P_j1] @ [V_j0;V_j1]: 16 full-K WMMAs for 256 columns --------
    const int jj = lh ? j1 : j0;               // lanes 0-15: K=0..15 rows (V_j0)
#pragma unroll                                 // lanes 16-31: K=16..31 rows (V_j1)
    for (int nt = 0; nt < 16; nt++) {
      BfVec bv;
      if (lh == 0 || have2) {
        const unsigned short* vp = vbT + (size_t)(cb + nt * 16 + nl) * HW + jj * 16;
#pragma unroll
        for (int i = 0; i < 16; i++) bv.s[i] = vp[i];
      } else {                                 // phantom tile -> zero (P2==0 too)
#pragma unroll
        for (int i = 0; i < 16; i++) bv.s[i] = 0;
      }
      acc[nt] = __builtin_amdgcn_wmma_f32_16x16x32_bf16(false, ap.v, false, bv.v,
                                                        (short)0, acc[nt], false, false);
    }

    if (u + 1 < NPAIR)                 // warm next pair's V^T rows
      __builtin_prefetch(vbT + (size_t)(cb + lane * 8) * HW + (j0 + 8) * 16, 0, 0);
  }

  // ---- merge the 4 key-group partials --------------------------------------
  if (h == 0 && nl == 0) {
#pragma unroll
    for (int r = 0; r < 8; r++) {
      m_s[g][r + 8 * lh] = mst[r];
      l_s[g][r + 8 * lh] = lst[r];
    }
  }
  __syncthreads();

  float wfac[8];
#pragma unroll
  for (int r = 0; r < 8; r++) {
    const int row = r + 8 * lh;
    float M = m_s[0][row];
    M = fmaxf(M, m_s[1][row]); M = fmaxf(M, m_s[2][row]); M = fmaxf(M, m_s[3][row]);
    float L = l_s[0][row] * __expf(m_s[0][row] - M)
            + l_s[1][row] * __expf(m_s[1][row] - M)
            + l_s[2][row] * __expf(m_s[2][row] - M)
            + l_s[3][row] * __expf(m_s[3][row] - M);
    wfac[r] = __expf(mst[r] - M) / L;   // includes 1/L normalization
  }

#pragma unroll
  for (int nt = 0; nt < 16; nt++) {
    F8 ta; ta.v = acc[nt];
#pragma unroll
    for (int r = 0; r < 8; r++)
      atomicAdd(&ctx_s[(r + 8 * lh) * CCH + cb + nt * 16 + nl], ta.f[r] * wfac[r]);
  }
  __syncthreads();

  const float gm = gamma[0];
  for (int i = threadIdx.x; i < 16 * CCH; i += 256) {
    const int row = i >> 9, col = i & 511;
    const size_t gi = (size_t)(q0 + row) * CCH + col;
    out[gi] = gm * ctx_s[i] + x[gi];
  }
}

// ---------------------------------------------------------------------------
extern "C" void kernel_launch(void* const* d_in, const int* in_sizes, int n_in,
                              void* d_out, int out_size, void* d_ws, size_t ws_size,
                              hipStream_t stream) {
  (void)in_sizes; (void)n_in; (void)out_size; (void)ws_size;
  const float* x     = (const float*)d_in[0];   // [1,84,112,512]
  const float* Wb    = (const float*)d_in[1];   // [512,64]
  const float* Wc    = (const float*)d_in[2];   // [512,64]
  const float* Wd    = (const float*)d_in[3];   // [512,512]
  const float* gamma = (const float*)d_in[4];   // [1]
  float* out = (float*)d_out;

  // Workspace: Q bf16 [9408x64] | K bf16 [9408x64] | V^T bf16 [512x9408]
  unsigned short* qb  = (unsigned short*)d_ws;
  unsigned short* kbf = qb  + (size_t)HW * 64;
  unsigned short* vbT = kbf + (size_t)HW * 64;

  pam_proj_kernel<<<(NT_J * NT_PROJ) / 8, 256, 0, stream>>>(x, Wb, Wc, Wd, qb, kbf, vbT);
  pam_attn_kernel<<<NT_J, 256, 0, stream>>>(x, qb, kbf, vbT, gamma, out);
}
